// FASTLoss_16621523436385
// MI455X (gfx1250) — compile-verified, matrix operand order checked by reading, not run on hardware
//
#include <hip/hip_runtime.h>
#include <hip/hip_bf16.h>
#include <stdint.h>

// ---------------------------------------------------------------------------
// FAST text-detection loss for MI455X (gfx1250, wave32).
// Pipeline: hmax -> vmax+sigmoid+histogram -> OHEM threshold -> dice(text)
//           -> dice(kernels) [TDM-staged] -> emb segment sums -> emb var pass
//           -> emb finalize [WMMA 16x16x4 Gram of means] -> combine.
// ---------------------------------------------------------------------------

#define Bn    8
#define Wd    640
#define Hh    640
#define HW    409600L          // 640*640
#define CH10  4096000L         // 10*HW
#define NPX   3276800L         // B*HW
#define NBINS 2048
#define KK    5                // NUM_KERNELS-1
#define TDM_CH 4096            // elements per TDM tile (16 KB f32)

typedef __attribute__((ext_vector_type(2)))  float    v2f;
typedef __attribute__((ext_vector_type(8)))  float    v8f;
typedef __attribute__((ext_vector_type(16))) _Float16 v16h;
typedef __attribute__((ext_vector_type(4)))  unsigned v4u;
typedef __attribute__((ext_vector_type(8)))  int      v8i_;
typedef __attribute__((ext_vector_type(4)))  int      v4i_;

__device__ __forceinline__ float sigf(float x) { return 1.f / (1.f + __expf(-x)); }

__device__ __forceinline__ float waveReduce(float v) {
  #pragma unroll
  for (int o = 16; o > 0; o >>= 1) v += __shfl_down(v, o, 32);
  return v;
}

// valid on thread 0; sbuf must hold >= 8 floats
__device__ __forceinline__ float blockReduce(float v, float* sbuf) {
  __syncthreads();
  int lane = threadIdx.x & 31, w = threadIdx.x >> 5;
  v = waveReduce(v);
  if (lane == 0) sbuf[w] = v;
  __syncthreads();
  int nw = blockDim.x >> 5;
  v = (threadIdx.x < (unsigned)nw) ? sbuf[threadIdx.x] : 0.f;
  if (w == 0) v = waveReduce(v);
  return v;
}

// --- K1: horizontal 9-tap running max of pred channel 0 (raw logits) -------
__global__ void k_hmax(const float* __restrict__ pred, float* __restrict__ rowmax) {
  long idx = (long)blockIdx.x * blockDim.x + threadIdx.x;
  if (idx >= NPX) return;
  long b = idx / HW, r = idx % HW;
  int y = (int)(r / Wd), x = (int)(r % Wd);
  const float* row = pred + b * CH10 + (long)y * Wd;
  int x0 = max(x - 4, 0), x1 = min(x + 4, Wd - 1);
  float mv = -3.402823466e38f;
  for (int xx = x0; xx <= x1; ++xx) mv = fmaxf(mv, row[xx]);
  rowmax[idx] = mv;
}

// --- K2: vertical 9-tap max, sigmoid -> dil; per-batch neg histogram -------
__global__ void k_vmax_hist(const float* __restrict__ rowmax,
                            const float* __restrict__ gt, const float* __restrict__ tmk,
                            float* __restrict__ dil, unsigned* __restrict__ hist,
                            unsigned* __restrict__ posneg) {
  __shared__ unsigned shist[NBINS];
  __shared__ unsigned sp, sn;
  for (int i = threadIdx.x; i < NBINS; i += blockDim.x) shist[i] = 0u;
  if (threadIdx.x == 0) { sp = 0u; sn = 0u; }
  __syncthreads();

  int b = blockIdx.x / 1600;                       // 1600 blocks x 256 px per batch
  long r = ((long)(blockIdx.x % 1600)) * 256 + threadIdx.x;
  long idx = (long)b * HW + r;
  int y = (int)(r / Wd), x = (int)(r % Wd);
  long base = (long)b * HW + x;
  int y0 = max(y - 4, 0), y1 = min(y + 4, Hh - 1);
  float mv = -3.402823466e38f;
  for (int yy = y0; yy <= y1; ++yy) mv = fmaxf(mv, rowmax[base + (long)yy * Wd]);
  float dv = sigf(mv);
  dil[idx] = dv;
  float g = gt[idx], m = tmk[idx];
  if (m > 0.5f) {
    if (g > 0.5f) atomicAdd(&sp, 1u);
    else {
      atomicAdd(&sn, 1u);
      int bin = min(NBINS - 1, (int)(dv * (float)NBINS));
      atomicAdd(&shist[bin], 1u);
    }
  }
  __syncthreads();
  for (int i = threadIdx.x; i < NBINS; i += blockDim.x)
    if (shist[i]) atomicAdd(&hist[b * NBINS + i], shist[i]);
  if (threadIdx.x == 0) {
    if (sp) atomicAdd(&posneg[b * 2 + 0], sp);
    if (sn) atomicAdd(&posneg[b * 2 + 1], sn);
  }
}

// --- K3: histogram-select OHEM threshold (replaces argsort) ----------------
__global__ void k_thresh(const unsigned* __restrict__ hist,
                         const unsigned* __restrict__ posneg, float* __restrict__ thr) {
  int b = threadIdx.x;
  if (b >= Bn) return;
  unsigned npos = posneg[b * 2 + 0], nnegT = posneg[b * 2 + 1];
  unsigned nneg = min(3u * npos, nnegT);
  float t = 2.f;                                   // select no negatives
  if (nneg > 0u) {
    unsigned cum = 0u;
    int bin = NBINS - 1;
    for (; bin >= 0; --bin) { cum += hist[b * NBINS + bin]; if (cum >= nneg) break; }
    if (bin < 0) bin = 0;
    t = (float)bin / (float)NBINS;                 // lower edge of threshold bin
  }
  thr[b] = t;
}

// --- K4: OHEM dice sums for text head --------------------------------------
__global__ void k_dice_text(const float* __restrict__ dil, const float* __restrict__ gt,
                            const float* __restrict__ tmk, const float* __restrict__ thr,
                            float* __restrict__ diceText) {
  __shared__ float sred[8];
  int b = blockIdx.x / 400;                        // 400 blocks x 1024 px per batch
  long r = ((long)(blockIdx.x % 400)) * 1024 + (long)threadIdx.x * 4;
  long idx = (long)b * HW + r;
  float th = thr[b];
  float4 d4 = *(const float4*)(dil + idx);
  float4 g4 = *(const float4*)(gt + idx);
  float4 m4 = *(const float4*)(tmk + idx);
  float I = 0.f, P = 0.f, T = 0.f;
  #pragma unroll
  for (int j = 0; j < 4; ++j) {
    float d = (&d4.x)[j], g = (&g4.x)[j], m = (&m4.x)[j];
    float msk = 0.f;
    if (m > 0.5f) msk = (g > 0.5f) ? 1.f : ((d >= th) ? 1.f : 0.f);
    I += msk * d * g; P += msk * d * d; T += msk * g * g;
  }
  float rI = blockReduce(I, sred);
  float rP = blockReduce(P, sred);
  float rT = blockReduce(T, sred);
  if (threadIdx.x == 0) {
    atomicAdd(&diceText[b * 3 + 0], rI);
    atomicAdd(&diceText[b * 3 + 1], rP);
    atomicAdd(&diceText[b * 3 + 2], rT);
  }
}

// --- K5: kernels dice; pred plane staged into LDS via Tensor Data Mover ----
__global__ void k_dice_kernels(const float* __restrict__ pred, const float* __restrict__ gtk,
                               const float* __restrict__ tmk, float* __restrict__ diceKern) {
  __shared__ float tile[TDM_CH];                   // first LDS object -> offset 0
  __shared__ float sred[8];
  int plane = blockIdx.x / 100;                    // 100 chunks of 4096 per plane
  int chunk = blockIdx.x % 100;
  int b = plane / KK, k = plane % KK;
  long pxBase = (long)chunk * TDM_CH;
  const float* src = pred + ((long)b * 10 + 1 + k) * HW + pxBase;

#if __has_builtin(__builtin_amdgcn_tensor_load_to_lds)
  if (threadIdx.x == 0) {
    unsigned long long ga = (unsigned long long)(uintptr_t)src;
    // D# group 0: count=1 valid, lds_addr=0, global_addr, type=2
    v4u g0;
    g0.x = 1u;
    g0.y = 0u;
    g0.z = (unsigned)(ga & 0xFFFFFFFFu);
    g0.w = (unsigned)((ga >> 32) & 0x01FFFFFFu) | (2u << 30);
    // D# group 1: data_size=4B; 1-D tile of TDM_CH elements
    v8i_ g1;
    g1[0] = 0x00020000;                            // data_size=2 (4B), no flags
    g1[1] = (int)(TDM_CH << 16);                   // tensor_dim0 lo16 @[31:16]
    g1[2] = (int)(1 << 16);                        // tensor_dim0 hi=0 | tensor_dim1=1
    g1[3] = (int)(TDM_CH << 16);                   // tensor_dim1 hi=0 | tile_dim0
    g1[4] = 1;                                     // tile_dim1=1, tile_dim2=0
    g1[5] = TDM_CH;                                // tensor_dim0_stride lo32
    g1[6] = (int)(TDM_CH << 16);                   // stride0 hi=0 | stride1 lo16
    g1[7] = 0;                                     // stride1 hi
    v4i_ z4 = {0, 0, 0, 0};
    v8i_ z8 = {0, 0, 0, 0, 0, 0, 0, 0};
    // clang-23 / therock-10.0 form: (g0, g1, g2, g3, g_extra, cpol)
    __builtin_amdgcn_tensor_load_to_lds(g0, g1, z4, z4, z8, 0);
    __builtin_amdgcn_s_wait_tensorcnt((short)0);
  }
#else
  for (int i = threadIdx.x; i < TDM_CH; i += blockDim.x) tile[i] = src[i];
#endif
  __syncthreads();

  const float* gk = gtk + ((long)b * KK + k) * HW + pxBase;
  const float* tb = tmk + (long)b * HW + pxBase;
  float I = 0.f, P = 0.f, T = 0.f;
  for (int i = threadIdx.x; i < TDM_CH; i += blockDim.x) {
    float p = sigf(tile[i]);
    float g = gk[i], m = tb[i];
    p *= m; g *= m;
    I += p * g; P += p * p; T += g * g;
  }
  float rI = blockReduce(I, sred);
  float rP = blockReduce(P, sred);
  float rT = blockReduce(T, sred);
  if (threadIdx.x == 0) {
    int row = b * KK + k;
    atomicAdd(&diceKern[row * 3 + 0], rI);
    atomicAdd(&diceKern[row * 3 + 1], rP);
    atomicAdd(&diceKern[row * 3 + 2], rT);
  }
}

// --- K6: embedding segment sums (16 labels x [4 sums + count]) -------------
__global__ void k_emb_sums(const float* __restrict__ pred, const int* __restrict__ inst,
                           const float* __restrict__ tmk, float* __restrict__ embSums) {
  __shared__ float ssum[16 * 5];
  for (int i = threadIdx.x; i < 80; i += blockDim.x) ssum[i] = 0.f;
  __syncthreads();
  int b = blockIdx.x / 800;                        // 800 blocks x 512 px per batch
  long r0 = ((long)(blockIdx.x % 800)) * 512 + threadIdx.x;
  #pragma unroll
  for (int it = 0; it < 2; ++it) {
    long r = r0 + it * 256;
    long idx = (long)b * HW + r;
    float m = tmk[idx];
    if (m > 0.f) {
      int g = inst[idx] & 15;
      const float* eb = pred + (long)b * CH10 + 6L * HW + r;
      atomicAdd(&ssum[g * 5 + 0], eb[0]);
      atomicAdd(&ssum[g * 5 + 1], eb[HW]);
      atomicAdd(&ssum[g * 5 + 2], eb[2 * HW]);
      atomicAdd(&ssum[g * 5 + 3], eb[3 * HW]);
      atomicAdd(&ssum[g * 5 + 4], 1.f);
    }
  }
  __syncthreads();
  for (int i = threadIdx.x; i < 80; i += blockDim.x)
    if (ssum[i] != 0.f) atomicAdd(&embSums[b * 80 + i], ssum[i]);
}

// --- K7: pull-term variance sums  cv = max(||e-mean||-0.5, 0)^2 ------------
__global__ void k_emb_cv(const float* __restrict__ pred, const int* __restrict__ inst,
                         const float* __restrict__ tmk, const float* __restrict__ embSums,
                         float* __restrict__ cvSums) {
  __shared__ float smean[16 * 4];
  __shared__ float scv[16];
  int b = blockIdx.x / 800;
  if (threadIdx.x < 64) {
    int l = threadIdx.x >> 2, d = threadIdx.x & 3;
    float c = embSums[b * 80 + l * 5 + 4];
    smean[l * 4 + d] = embSums[b * 80 + l * 5 + d] / fmaxf(c, 1.f);
  }
  if (threadIdx.x < 16) scv[threadIdx.x] = 0.f;
  __syncthreads();
  long r0 = ((long)(blockIdx.x % 800)) * 512 + threadIdx.x;
  #pragma unroll
  for (int it = 0; it < 2; ++it) {
    long r = r0 + it * 256;
    long idx = (long)b * HW + r;
    float m = tmk[idx];
    if (m > 0.f) {
      int g = inst[idx] & 15;
      const float* eb = pred + (long)b * CH10 + 6L * HW + r;
      float d0 = eb[0] - smean[g * 4 + 0];
      float d1 = eb[HW] - smean[g * 4 + 1];
      float d2 = eb[2 * HW] - smean[g * 4 + 2];
      float d3 = eb[3 * HW] - smean[g * 4 + 3];
      float dist = sqrtf(d0 * d0 + d1 * d1 + d2 * d2 + d3 * d3 + 1e-12f);
      float cv = fmaxf(dist - 0.5f, 0.f);
      atomicAdd(&scv[g], cv * cv);
    }
  }
  __syncthreads();
  if (threadIdx.x < 16 && scv[threadIdx.x] != 0.f)
    atomicAdd(&cvSums[b * 16 + threadIdx.x], scv[threadIdx.x]);
}

// --- K8: per-batch lv/ld/lr; 16x16 Gram of means via V_WMMA_F32_16X16X4_F32
__global__ void k_emb_finalize(const float* __restrict__ embSums,
                               const float* __restrict__ cvSums,
                               float* __restrict__ lvldlr) {
  int b = blockIdx.x;
  int lane = threadIdx.x;                          // one full wave32
  int l = lane & 15;
  const float* S = embSums + (b * 16 + l) * 5;
  float cnt = S[4];
  float cdiv = fmaxf(cnt, 1.f);
  float m0 = S[0] / cdiv, m1 = S[1] / cdiv, m2 = S[2] / cdiv, m3 = S[3] / cdiv;
  bool present = cnt > 0.f;
  float nj = m0 * m0 + m1 * m1 + m2 * m2 + m3 * m3;  // ||mean_l||^2

  unsigned pm = (unsigned)(__ballot(present) & 0xFFFFull);
  float nf = (float)__popc(pm);

  // Gram = M (16x4) x M^T (4x16). A-layout (f32 16x4): VGPR0 = K0|K2, VGPR1 = K1|K3.
  v8f c = {0.f, 0.f, 0.f, 0.f, 0.f, 0.f, 0.f, 0.f};
#if __has_builtin(__builtin_amdgcn_wmma_f32_16x16x4_f32)
  v2f a;
  a.x = (lane < 16) ? m0 : m2;
  a.y = (lane < 16) ? m1 : m3;
  c = __builtin_amdgcn_wmma_f32_16x16x4_f32(false, a, false, a, (short)0, c,
                                            false, false);
#else
  // fallback: f16 WMMA (codegen-confirmed); means in K=0..3, zero padded
  v16h a = (v16h)(_Float16)0.f;
  v16h bb = (v16h)(_Float16)0.f;
  if (lane < 16) {
    a[0] = (_Float16)m0; a[1] = (_Float16)m1; a[2] = (_Float16)m2; a[3] = (_Float16)m3;
    bb[0] = (_Float16)m0; bb[1] = (_Float16)m1; bb[2] = (_Float16)m2; bb[3] = (_Float16)m3;
  }
  c = __builtin_amdgcn_wmma_f32_16x16x32_f16(false, a, false, bb, (short)0, c,
                                             false, false);
#endif

  // ld: sum over i<j present of max(3 - pd, 0)^2 ; pd^2 = n_i + n_j - 2*Gram_ij
  int half = lane >> 4;
  float ldp = 0.f;
  #pragma unroll
  for (int r = 0; r < 8; ++r) {
    int M = r + 8 * half;                          // C layout: VGPR r -> M=r / M=r+8
    int N = l;
    float gij = c[r];
    float niM = __shfl(nj, M, 32);
    bool ok = ((pm >> M) & 1u) && ((pm >> N) & 1u) && (M < N);
    float pd = sqrtf(fmaxf(niM + nj - 2.f * gij, 0.f) + 1e-12f);
    float t = fmaxf(3.0f - pd, 0.f);               // 2*DELTA_D = 3.0
    ldp += ok ? t * t : 0.f;
  }
  float ld = waveReduce(ldp) / fmaxf(nf * (nf - 1.f), 1.f);

  float lvl = 0.f;
  if (lane < 16) {
    float per = cvSums[b * 16 + l] / fmaxf(cnt, 1.f);
    lvl = present ? per : 0.f;
  }
  float lv = waveReduce(lvl) / fmaxf(nf, 1.f);

  float lrl = (lane < 16 && present) ? sqrtf(nj + 1e-12f) : 0.f;
  float lr = waveReduce(lrl) / fmaxf(nf, 1.f);

  if (lane == 0) {
    float act = (nf > 1.f) ? 1.f : 0.f;
    lvldlr[b * 3 + 0] = act * lv;
    lvldlr[b * 3 + 1] = act * ld;
    lvldlr[b * 3 + 2] = act * lr;
  }
}

// --- K9: combine everything into the 4 output scalars ----------------------
__global__ void k_finalize(const float* __restrict__ diceText,
                           const float* __restrict__ diceKern,
                           const float* __restrict__ lvldlr, float* __restrict__ out) {
  if (threadIdx.x != 0 || blockIdx.x != 0) return;
  float lt = 0.f;
  for (int b = 0; b < Bn; ++b) {
    float I = diceText[b * 3], P = diceText[b * 3 + 1], T = diceText[b * 3 + 2];
    lt += 1.f - 2.f * I / (P + T + 1e-6f);
  }
  lt /= (float)Bn;
  float lk = 0.f;
  for (int r = 0; r < Bn * KK; ++r) {
    float I = diceKern[r * 3], P = diceKern[r * 3 + 1], T = diceKern[r * 3 + 2];
    lk += 1.f - 2.f * I / (P + T + 1e-6f);
  }
  lk /= (float)(Bn * KK);
  float lv = 0.f, ld = 0.f, lr = 0.f;
  for (int b = 0; b < Bn; ++b) {
    lv += lvldlr[b * 3 + 0];
    ld += lvldlr[b * 3 + 1];
    lr += lvldlr[b * 3 + 2];
  }
  float le = 0.25f * (lv + ld + 0.001f * (lr / (float)Bn));
  float loss = lk + 0.5f * lt + le;
  out[0] = loss; out[1] = lt; out[2] = lk; out[3] = le;
}

// ---------------------------------------------------------------------------
extern "C" void kernel_launch(void* const* d_in, const int* in_sizes, int n_in,
                              void* d_out, int out_size, void* d_ws, size_t ws_size,
                              hipStream_t stream) {
  const float* pred = (const float*)d_in[0];
  const float* gt_text = (const float*)d_in[1];
  const float* gt_kernels = (const float*)d_in[2];
  const float* tmask = (const float*)d_in[3];
  const int* gt_instance = (const int*)d_in[4];
  float* out = (float*)d_out;

  char* ws = (char*)d_ws;
  // workspace layout (bytes)
  const size_t OFF_ROWMAX = 0;                       // NPX f32 = 13,107,200
  const size_t OFF_DIL = 13107200;                   // NPX f32 = 13,107,200
  const size_t OFF_ACC = 26214400;                   // accumulators (zeroed)
  const size_t OFF_HIST = OFF_ACC;                   // 8*2048 u32
  const size_t OFF_PN = OFF_HIST + 65536;            // 16 u32
  const size_t OFF_THR = OFF_PN + 64;                // 8 f32
  const size_t OFF_DT = OFF_THR + 32;                // 24 f32
  const size_t OFF_DK = OFF_DT + 96;                 // 120 f32
  const size_t OFF_ES = OFF_DK + 480;                // 640 f32
  const size_t OFF_CV = OFF_ES + 2560;               // 128 f32
  const size_t OFF_LV = OFF_CV + 512;                // 24 f32
  const size_t ACC_BYTES = (OFF_LV + 96) - OFF_ACC;

  float* rowmax = (float*)(ws + OFF_ROWMAX);
  float* dil = (float*)(ws + OFF_DIL);
  unsigned* hist = (unsigned*)(ws + OFF_HIST);
  unsigned* posneg = (unsigned*)(ws + OFF_PN);
  float* thr = (float*)(ws + OFF_THR);
  float* diceText = (float*)(ws + OFF_DT);
  float* diceKern = (float*)(ws + OFF_DK);
  float* embSums = (float*)(ws + OFF_ES);
  float* cvSums = (float*)(ws + OFF_CV);
  float* lvldlr = (float*)(ws + OFF_LV);

  (void)hipMemsetAsync(ws + OFF_ACC, 0, ACC_BYTES, stream);

  k_hmax<<<(int)(NPX / 256), 256, 0, stream>>>(pred, rowmax);
  k_vmax_hist<<<Bn * 1600, 256, 0, stream>>>(rowmax, gt_text, tmask, dil, hist, posneg);
  k_thresh<<<1, 32, 0, stream>>>(hist, posneg, thr);
  k_dice_text<<<Bn * 400, 256, 0, stream>>>(dil, gt_text, tmask, thr, diceText);
  k_dice_kernels<<<Bn * KK * 100, 256, 0, stream>>>(pred, gt_kernels, tmask, diceKern);
  k_emb_sums<<<Bn * 800, 256, 0, stream>>>(pred, gt_instance, tmask, embSums);
  k_emb_cv<<<Bn * 800, 256, 0, stream>>>(pred, gt_instance, tmask, embSums, cvSums);
  k_emb_finalize<<<Bn, 32, 0, stream>>>(embSums, cvSums, lvldlr);
  k_finalize<<<1, 32, 0, stream>>>(diceText, diceKern, lvldlr, out);
}
